// SelfAttention_40896678593195
// MI455X (gfx1250) — matrix operand order repaired and checked by reading, not compiled, not run on hardware
//
#include <hip/hip_runtime.h>

#define BATCH 8
#define CDIM  256
#define NPOS  4096      // 4*32*32
#define QKD   32
#define NCHUNK (NPOS / 32)
#define LN_EPS 1e-5f

typedef __bf16 v16bf __attribute__((ext_vector_type(16)));
typedef __bf16 v8bf  __attribute__((ext_vector_type(8)));
typedef float  v8f   __attribute__((ext_vector_type(8)));
typedef int    v4i   __attribute__((ext_vector_type(4)));

static __device__ __forceinline__ v8f wmma_bf16(v16bf a, v16bf b, v8f c) {
  // D = A(16x32 bf16) * B(32x16 bf16) + C(16x16 f32)
  return __builtin_amdgcn_wmma_f32_16x16x32_bf16(false, a, false, b, (short)0, c, false, false);
}

static __device__ __forceinline__ v16bf concat8(v8bf lo, v8bf hi) {
  return __builtin_shufflevector(lo, hi, 0,1,2,3,4,5,6,7,8,9,10,11,12,13,14,15);
}

// ---------------------------------------------------------------------------
// gfx1250 async global->LDS copy (ASYNCcnt path), with safe fallback.
// Builtin signature (from hipcc diagnostic): (v4i AS1*, v4i AS3*, Ii, Ii).
// ---------------------------------------------------------------------------
#if defined(__gfx1250__) && __has_builtin(__builtin_amdgcn_global_load_async_to_lds_b128) && \
    __has_builtin(__builtin_amdgcn_s_wait_asynccnt)
#define USE_ASYNC 1
#else
#define USE_ASYNC 0
#endif

static __device__ __forceinline__ void ldsCopy16B(const __bf16* g, __bf16* l) {
#if USE_ASYNC
  __builtin_amdgcn_global_load_async_to_lds_b128(
      (__attribute__((address_space(1))) v4i*)(void*)g,
      (__attribute__((address_space(3))) v4i*)(void*)l, 0, 0);
#else
  *(v8bf*)l = *(const v8bf*)g;
#endif
}
static __device__ __forceinline__ void asyncWait0() {
#if USE_ASYNC
  __builtin_amdgcn_s_wait_asynccnt(0);
#endif
}
static __device__ __forceinline__ void asyncWait4() {
#if USE_ASYNC
  __builtin_amdgcn_s_wait_asynccnt(4);   // keep only the newest prefetch group in flight
#endif
}

// ---------------------------------------------------------------------------
// Kernel 0: zero LayerNorm accumulators
// ---------------------------------------------------------------------------
__global__ void init_kernel(float* lnAcc) {
  if (threadIdx.x < 2 * BATCH) lnAcc[threadIdx.x] = 0.0f;
}

// ---------------------------------------------------------------------------
// Kernel 0b: f32 -> bf16 weight conversion (W1, W2 stay L2-resident as bf16)
// ---------------------------------------------------------------------------
__global__ void cvtbf_kernel(const float* __restrict__ in, __bf16* __restrict__ out, int n) {
  int i = blockIdx.x * 256 + threadIdx.x;
  if (i < n) out[i] = (__bf16)in[i];
}

// ---------------------------------------------------------------------------
// Kernel 1: pointwise projection  out[o,n] = (sum_c W[o,c]*x[b,c,n] + bias[o])*scale
//   MODE 0: row-major [B, N, Co] (Q, K);  MODE 1: col-major [B, Co, N] (V^T)
// ---------------------------------------------------------------------------
template <int MODE>
__global__ void proj_kernel(const float* __restrict__ x, const float* __restrict__ W,
                            const float* __restrict__ bias, __bf16* __restrict__ out,
                            int Co, float scale) {
  const int lane = threadIdx.x & 31;
  const int col  = lane & 15;
  const int hl   = lane >> 4;
  const int nt = blockIdx.x, ot = blockIdx.y, b = blockIdx.z;
  const int n    = nt * 16 + col;
  const int orow = ot * 16 + col;

  const float* wrow = W + (size_t)orow * CDIM;
  const float* xb   = x + (size_t)b * CDIM * NPOS;

  v8f acc = {};
  for (int c0 = 0; c0 < CDIM; c0 += 32) {
    v16bf a, bm;
#pragma unroll
    for (int i = 0; i < 8; ++i) {            // A layout: K = hl*16+i , hl*16+8+i
      a[i]     = (__bf16)wrow[c0 + hl * 16 + i];
      a[8 + i] = (__bf16)wrow[c0 + hl * 16 + 8 + i];
    }
#pragma unroll
    for (int e = 0; e < 16; ++e)             // B layout: K = hl*16+e, col = n
      bm[e] = (__bf16)xb[(size_t)(c0 + hl * 16 + e) * NPOS + n];
    acc = wmma_bf16(a, bm, acc);
  }
#pragma unroll
  for (int j = 0; j < 8; ++j) {              // D: row o = ot*16+hl*8+j, col = n
    int o = ot * 16 + hl * 8 + j;
    float v = (acc[j] + bias[o]) * scale;
    if (MODE == 1) out[((size_t)b * Co + o) * NPOS + n] = (__bf16)v;
    else           out[((size_t)b * NPOS + n) * Co + o] = (__bf16)v;
  }
}

// ---------------------------------------------------------------------------
// Kernel 2: per-column softmax stats (softmax is over the QUERY axis m).
// 8 waves split the 256 m-tiles; online max/sum; shfl + LDS combine.
// ---------------------------------------------------------------------------
__global__ void colstats_kernel(const __bf16* __restrict__ Q, const __bf16* __restrict__ K,
                                float* __restrict__ colmax, float* __restrict__ colsum) {
  const int tid = threadIdx.x;
  const int lane = tid & 31, wv = tid >> 5;
  const int col = lane & 15, hl = lane >> 4;
  const int nt = blockIdx.x, b = blockIdx.y;
  const int n = nt * 16 + col;

  v16bf kb = *(const v16bf*)(K + ((size_t)b * NPOS + n) * QKD + hl * 16);

  float runm = -1e30f, runs = 0.0f;
  for (int mt = wv; mt < NPOS / 16; mt += 8) {
    const __bf16* qrow = Q + ((size_t)b * NPOS + mt * 16 + col) * QKD;
    v16bf a = concat8(*(const v8bf*)(qrow + hl * 16),
                      *(const v8bf*)(qrow + hl * 16 + 8));
    v8f z = {};
    v8f d = wmma_bf16(a, kb, z);
    float tm = d[0];
#pragma unroll
    for (int j = 1; j < 8; ++j) tm = fmaxf(tm, d[j]);
    float nm = fmaxf(runm, tm);
    float s = runs * __expf(runm - nm);
#pragma unroll
    for (int j = 0; j < 8; ++j) s += __expf(d[j] - nm);
    runm = nm; runs = s;
  }
  float om = __shfl_xor(runm, 16, 32);
  float os = __shfl_xor(runs, 16, 32);
  float nm = fmaxf(runm, om);
  float ns = runs * __expf(runm - nm) + os * __expf(om - nm);

  __shared__ float smax[8][16];
  __shared__ float ssum[8][16];
  if (lane < 16) { smax[wv][col] = nm; ssum[wv][col] = ns; }
  __syncthreads();
  if (tid < 16) {
    float M = smax[0][tid];
    for (int w2 = 1; w2 < 8; ++w2) M = fmaxf(M, smax[w2][tid]);
    float S = 0.0f;
    for (int w2 = 0; w2 < 8; ++w2) S += ssum[w2][tid] * __expf(smax[w2][tid] - M);
    colmax[(size_t)b * NPOS + nt * 16 + tid] = M;
    colsum[(size_t)b * NPOS + nt * 16 + tid] = S;
  }
}

// ---------------------------------------------------------------------------
// Kernel 3: out[m,c] = sum_n softmax_col(logits)[m,n] * V[n,c].
// Block = 8 waves, 64 query rows (4 m-tiles). Wave pair (p = w>>1) shares one
// m-tile; even wave computes the logits tile + softmax staging, both halves
// consume it. V chunks (32n x 256c) are double-buffered in LDS via async
// global->LDS copies (one row per thread, 4x b128), so V is read from HBM/L2
// once per block instead of once per wave.
// ---------------------------------------------------------------------------
__global__ void attn_kernel(const __bf16* __restrict__ Q, const __bf16* __restrict__ K,
                            const __bf16* __restrict__ Vt,
                            const float* __restrict__ colmax, const float* __restrict__ colsum,
                            __bf16* __restrict__ attnOut) {
  const int tid = threadIdx.x;
  const int lane = tid & 31, w = tid >> 5;     // 8 waves
  const int col = lane & 15, hl = lane >> 4;
  const int p  = w >> 1;                       // local m-tile 0..3
  const int ch = w & 1;                        // channel half 0..1
  const int b = blockIdx.y;
  const int mbase = blockIdx.x * 64;

  __shared__ __align__(32) __bf16 Vbuf[2][CDIM][32]; // 32 KB double buffer
  __shared__ __align__(32) __bf16 stage[4][16][32];  // 4 KB softmax-weight tiles

  const __bf16* qrow = Q + ((size_t)b * NPOS + mbase + p * 16 + col) * QKD;
  v16bf qa = concat8(*(const v8bf*)(qrow + hl * 16),
                     *(const v8bf*)(qrow + hl * 16 + 8));

  const __bf16* vbase = Vt + (size_t)b * CDIM * NPOS;
  v8f acc[8];
#pragma unroll
  for (int ct = 0; ct < 8; ++ct) acc[ct] = (v8f){};

  // prologue: stage chunk 0 (thread tid copies V row c=tid, 64 bytes)
  {
    const __bf16* g = vbase + (size_t)tid * NPOS;
    __bf16* l = &Vbuf[0][tid][0];
#pragma unroll
    for (int u = 0; u < 4; ++u) ldsCopy16B(g + u * 8, l + u * 8);
  }

  for (int i = 0; i < NCHUNK; ++i) {
    const int n0 = i * 32;
    if (i + 1 < NCHUNK) {                      // prefetch next chunk, then wait
      const __bf16* g = vbase + (size_t)tid * NPOS + (n0 + 32);
      __bf16* l = &Vbuf[(i + 1) & 1][tid][0];
#pragma unroll
      for (int u = 0; u < 4; ++u) ldsCopy16B(g + u * 8, l + u * 8);
      asyncWait4();                            // chunk-i copies complete
    } else {
      asyncWait0();
    }

    if (ch == 0) {                             // one logits producer per pair
#pragma unroll
      for (int t = 0; t < 2; ++t) {
        int n = n0 + t * 16 + col;
        v16bf kb = *(const v16bf*)(K + ((size_t)b * NPOS + n) * QKD + hl * 16);
        v8f z = {};
        v8f d = wmma_bf16(qa, kb, z);
        float cm  = colmax[(size_t)b * NPOS + n];
        float inv = 1.0f / colsum[(size_t)b * NPOS + n];
#pragma unroll
        for (int j = 0; j < 8; ++j)            // D row m = hl*8+j, chunk col t*16+col
          stage[p][hl * 8 + j][t * 16 + col] = (__bf16)(__expf(d[j] - cm) * inv);
      }
    }
    __syncthreads();                           // Vbuf[i&1] + stage[p] ready

    const v8bf* sp = (const v8bf*)&stage[p][col][0];
    v16bf aa = concat8(sp[2 * hl], sp[2 * hl + 1]);    // A layout: row m = col
#pragma unroll
    for (int ct = 0; ct < 8; ++ct) {
      int cl = ch * 128 + ct * 16 + col;       // B operand from LDS: col=c, K(n)=hl*16+e
      v16bf vb = *(const v16bf*)&Vbuf[i & 1][cl][hl * 16];
      acc[ct] = wmma_bf16(aa, vb, acc[ct]);
    }
    __syncthreads();                           // protect stage + consumed V buffer
  }

#pragma unroll
  for (int ct = 0; ct < 8; ++ct) {
    int c = ch * 128 + ct * 16 + col;
#pragma unroll
    for (int j = 0; j < 8; ++j) {
      int m = mbase + p * 16 + hl * 8 + j;
      attnOut[((size_t)b * NPOS + m) * CDIM + c] = (__bf16)acc[ct][j];
    }
  }
}

// ---------------------------------------------------------------------------
// Kernel 4: per-position MLP (256 -> relu -> 256) + LayerNorm partial sums.
// bf16 weights; GEMM1 result lands in LDS [m][o] (that IS the repack).
// ---------------------------------------------------------------------------
__global__ void mlp_kernel(const __bf16* __restrict__ attnOut,
                           const __bf16* __restrict__ W1bf, const float* __restrict__ b1,
                           const __bf16* __restrict__ W2bf, const float* __restrict__ b2,
                           float* __restrict__ mlpOut, float* __restrict__ lnAcc) {
  const int tid = threadIdx.x;
  const int lane = tid & 31, wv = tid >> 5;    // 4 waves
  const int col = lane & 15, hl = lane >> 4;
  const int mt = blockIdx.x, b = blockIdx.y;

  __shared__ __align__(32) __bf16 h1[16][CDIM];

  const __bf16* arow = attnOut + ((size_t)b * NPOS + mt * 16 + col) * CDIM;

#pragma unroll
  for (int ot = 0; ot < 4; ++ot) {
    int o = (wv * 4 + ot) * 16 + col;
    const __bf16* w1row = W1bf + (size_t)o * CDIM;
    v8f acc = {};
    for (int c0 = 0; c0 < CDIM; c0 += 32) {
      v16bf a = concat8(*(const v8bf*)(arow + c0 + hl * 16),
                        *(const v8bf*)(arow + c0 + hl * 16 + 8));
      v16bf bm = *(const v16bf*)(w1row + c0 + hl * 16);
      acc = wmma_bf16(a, bm, acc);
    }
    float bias = b1[o];
#pragma unroll
    for (int j = 0; j < 8; ++j)
      h1[hl * 8 + j][o] = (__bf16)fmaxf(acc[j] + bias, 0.0f);
  }
  __syncthreads();

  float s1 = 0.0f, s2 = 0.0f;
#pragma unroll
  for (int ot = 0; ot < 4; ++ot) {
    int o = (wv * 4 + ot) * 16 + col;
    const __bf16* w2row = W2bf + (size_t)o * CDIM;
    v8f acc = {};
    for (int c0 = 0; c0 < CDIM; c0 += 32) {
      const v8bf* hp = (const v8bf*)&h1[col][c0];
      v16bf a = concat8(hp[2 * hl], hp[2 * hl + 1]);
      v16bf bm = *(const v16bf*)(w2row + c0 + hl * 16);
      acc = wmma_bf16(a, bm, acc);
    }
    float bias = b2[o];
#pragma unroll
    for (int j = 0; j < 8; ++j) {
      float v = acc[j] + bias;
      s1 += v; s2 += v * v;
      int m = mt * 16 + hl * 8 + j;
      mlpOut[((size_t)b * NPOS + m) * CDIM + o] = v;
    }
  }
  for (int off = 16; off > 0; off >>= 1) {
    s1 += __shfl_xor(s1, off, 32);
    s2 += __shfl_xor(s2, off, 32);
  }
  if (lane == 0) {
    atomicAdd(&lnAcc[b], s1);
    atomicAdd(&lnAcc[BATCH + b], s2);
  }
}

// ---------------------------------------------------------------------------
// Kernel 5: finalize LayerNorm, apply gamma/beta, write [B, C, Z, H, W].
// ---------------------------------------------------------------------------
__global__ void ln_kernel(const float* __restrict__ mlpOut, const float* __restrict__ lnAcc,
                          const float* __restrict__ gamma, const float* __restrict__ beta,
                          float* __restrict__ out) {
  size_t idx = (size_t)blockIdx.x * blockDim.x + threadIdx.x;
  const size_t total = (size_t)BATCH * CDIM * NPOS;
  if (idx >= total) return;
  int b = (int)(idx / ((size_t)CDIM * NPOS));
  size_t rem = idx % ((size_t)CDIM * NPOS);
  int c = (int)(rem / NPOS);
  int n = (int)(rem % NPOS);
  const float inv_count = 1.0f / (float)((size_t)CDIM * NPOS);
  float mu  = lnAcc[b] * inv_count;
  float var = lnAcc[BATCH + b] * inv_count - mu * mu;
  float v = mlpOut[((size_t)b * NPOS + n) * CDIM + c];
  float r = rsqrtf(var + LN_EPS);
  out[idx] = (v - mu) * r * gamma[rem] + beta[rem];
}

// ---------------------------------------------------------------------------
extern "C" void kernel_launch(void* const* d_in, const int* in_sizes, int n_in,
                              void* d_out, int out_size, void* d_ws, size_t ws_size,
                              hipStream_t stream) {
  (void)in_sizes; (void)n_in; (void)out_size; (void)ws_size;
  const float* x  = (const float*)d_in[0];
  const float* Wq = (const float*)d_in[1];
  const float* bq = (const float*)d_in[2];
  const float* Wk = (const float*)d_in[3];
  const float* bk = (const float*)d_in[4];
  const float* Wv = (const float*)d_in[5];
  const float* bv = (const float*)d_in[6];
  const float* W1 = (const float*)d_in[7];
  const float* b1 = (const float*)d_in[8];
  const float* W2 = (const float*)d_in[9];
  const float* b2 = (const float*)d_in[10];
  const float* gamma = (const float*)d_in[11];
  const float* beta  = (const float*)d_in[12];
  float* out = (float*)d_out;

  char* ws = (char*)d_ws;
  size_t o = 0;
  __bf16* Qbf = (__bf16*)(ws + o);    o += (size_t)BATCH * NPOS * QKD * 2;   // 2 MB
  __bf16* Kbf = (__bf16*)(ws + o);    o += (size_t)BATCH * NPOS * QKD * 2;   // 2 MB
  __bf16* Vt  = (__bf16*)(ws + o);    o += (size_t)BATCH * CDIM * NPOS * 2;  // 16 MB
  float* colmax = (float*)(ws + o);   o += (size_t)BATCH * NPOS * 4;         // 128 KB
  float* colsum = (float*)(ws + o);   o += (size_t)BATCH * NPOS * 4;         // 128 KB
  __bf16* attnO = (__bf16*)(ws + o);  o += (size_t)BATCH * NPOS * CDIM * 2;  // 16 MB
  float* mlpOut = (float*)(ws + o);   o += (size_t)BATCH * NPOS * CDIM * 4;  // 32 MB
  float* lnAcc  = (float*)(ws + o);   o += 256;                              // 64 B (+pad)
  __bf16* W1bf  = (__bf16*)(ws + o);  o += (size_t)CDIM * CDIM * 2;          // 128 KB
  __bf16* W2bf  = (__bf16*)(ws + o);                                         // 128 KB

  const float scale = 0.17677669529663687f;  // (DIM/8)^-0.5 = 32^-0.5

  init_kernel<<<1, 32, 0, stream>>>(lnAcc);
  cvtbf_kernel<<<dim3((CDIM * CDIM + 255) / 256), dim3(256), 0, stream>>>(W1, W1bf, CDIM * CDIM);
  cvtbf_kernel<<<dim3((CDIM * CDIM + 255) / 256), dim3(256), 0, stream>>>(W2, W2bf, CDIM * CDIM);
  proj_kernel<0><<<dim3(NPOS / 16, QKD / 16, BATCH), dim3(32), 0, stream>>>(
      x, Wq, bq, Qbf, QKD, scale);
  proj_kernel<0><<<dim3(NPOS / 16, QKD / 16, BATCH), dim3(32), 0, stream>>>(
      x, Wk, bk, Kbf, QKD, 1.0f);
  proj_kernel<1><<<dim3(NPOS / 16, CDIM / 16, BATCH), dim3(32), 0, stream>>>(
      x, Wv, bv, Vt, CDIM, 1.0f);
  colstats_kernel<<<dim3(NPOS / 16, BATCH), dim3(256), 0, stream>>>(
      Qbf, Kbf, colmax, colsum);
  attn_kernel<<<dim3(NPOS / 64, BATCH), dim3(256), 0, stream>>>(
      Qbf, Kbf, Vt, colmax, colsum, attnO);
  mlp_kernel<<<dim3(NPOS / 16, BATCH), dim3(128), 0, stream>>>(
      attnO, W1bf, b1, W2bf, b2, mlpOut, lnAcc);
  const int total = BATCH * CDIM * NPOS;
  ln_kernel<<<dim3((total + 255) / 256), dim3(256), 0, stream>>>(
      mlpOut, lnAcc, gamma, beta, out);
}
